// BiLSTMCRF_2276332667511
// MI455X (gfx1250) — matrix operand order repaired and checked by reading, not compiled, not run on hardware
//
#include <hip/hip_runtime.h>
#include <hip/hip_bf16.h>

// Problem constants
#define BB   128
#define TT   512              // power of two
#define DD   300
#define DP   320              // D padded to multiple of 32 (K of bf16 WMMA)
#define HH   150
#define HP   160              // H padded to multiple of 32
#define LL   25
#define NP   608              // padded gate width: 152 units * 4 gates (unit-major)
#define NTIL 38               // NP/16
#define ROWS (BB*TT)          // 65536

typedef __attribute__((ext_vector_type(16))) __bf16        v16bf;
typedef __attribute__((ext_vector_type(8)))  float         v8f;
typedef __attribute__((ext_vector_type(8)))  unsigned int  v8u;
typedef __attribute__((ext_vector_type(4)))  unsigned int  v4u;

__device__ __forceinline__ unsigned short f2bf(float f) {
  unsigned int u = __builtin_bit_cast(unsigned int, f);
  u += 0x7fffu + ((u >> 16) & 1u);            // round-to-nearest-even
  return (unsigned short)(u >> 16);
}
__device__ __forceinline__ float bf2f(unsigned int h16) {
  unsigned int u = (h16 & 0xffffu) << 16;
  return __builtin_bit_cast(float, u);
}
__device__ __forceinline__ float sigf(float x) { return 1.0f / (1.0f + __expf(-x)); }
__device__ __forceinline__ float tanh_fast(float x) {
  float e = __expf(2.0f * x);                 // hardware v_exp_f32
  return 1.0f - 2.0f / (e + 1.0f);
}

// ---------------------------------------------------------------------------
// 0) lengths[b] = sum(mask[b,:])
__global__ void k_lengths(const float* __restrict__ mask, int* __restrict__ len) {
  int b = threadIdx.x;
  if (b >= BB) return;
  float s = 0.f;
  for (int t = 0; t < TT; ++t) s += mask[b * TT + t];
  len[b] = (int)(s + 0.5f);
}

// 1) pack input to bf16, K padded to DP
__global__ void k_pack_x(const float* __restrict__ x, unsigned short* __restrict__ xbf) {
  long long i = (long long)blockIdx.x * blockDim.x + threadIdx.x;
  long long total = (long long)ROWS * DP;
  if (i >= total) return;
  int k = (int)(i % DP);
  long long r = i / DP;
  float v = (k < DD) ? x[r * DD + k] : 0.f;
  xbf[i] = f2bf(v);
}

// 2) pack weights to bf16, gate rows permuted unit-major: prow = u*4+g  (u<152 pad)
__global__ void k_pack_w(const float* Wih_f, const float* Whh_f,
                         const float* bih_f, const float* bhh_f,
                         const float* Wih_b, const float* Whh_b,
                         const float* bih_b, const float* bhh_b,
                         unsigned short* __restrict__ wih_p,
                         unsigned short* __restrict__ whh_p,
                         float* __restrict__ bias_p) {
  long long tid = (long long)blockIdx.x * blockDim.x + threadIdx.x;
  long long stride = (long long)gridDim.x * blockDim.x;
  for (long long i = tid; i < 2LL * NP * DP; i += stride) {
    int k = (int)(i % DP); long long rr = i / DP;
    int prow = (int)(rr % NP); int dir = (int)(rr / NP);
    int u = prow >> 2, g = prow & 3;
    const float* W = dir ? Wih_b : Wih_f;
    float v = (u < HH && k < DD) ? W[(g * HH + u) * DD + k] : 0.f;
    wih_p[i] = f2bf(v);
  }
  for (long long i = tid; i < 2LL * NP * HP; i += stride) {
    int k = (int)(i % HP); long long rr = i / HP;
    int prow = (int)(rr % NP); int dir = (int)(rr / NP);
    int u = prow >> 2, g = prow & 3;
    const float* W = dir ? Whh_b : Whh_f;
    float v = (u < HH && k < HH) ? W[(g * HH + u) * HH + k] : 0.f;
    whh_p[i] = f2bf(v);
  }
  for (long long i = tid; i < 2LL * NP; i += stride) {
    int prow = (int)(i % NP); int dir = (int)(i / NP);
    int u = prow >> 2, g = prow & 3;
    float v = 0.f;
    if (u < HH) {
      v = dir ? (bih_b[g * HH + u] + bhh_b[g * HH + u])
              : (bih_f[g * HH + u] + bhh_f[g * HH + u]);
    }
    bias_p[i] = v;
  }
}

// ---------------------------------------------------------------------------
// Gpre is stored in WMMA-C "tile register order": per (dir,t,bc,nt) tile,
// lane l keeps its 8 accumulators as 4 packed-bf16 dwords at lane*16 bytes.
// uint offset of a tile: (((dir*TT + t)*8 + bc)*NTIL + nt)*128
// ---------------------------------------------------------------------------
// 3) Input-side GEMM: grid = (TT*8, 2) -> blockIdx.x = t*8 + bc, 256 thr (8 waves)
__global__ void k_gemm_gpre(const unsigned short* __restrict__ xbf,
                            const unsigned short* __restrict__ wih_p,
                            const float* __restrict__ bias_p,
                            const int* __restrict__ len,
                            unsigned int* __restrict__ gpre_u) {
  const int t   = blockIdx.x >> 3;
  const int bc  = blockIdx.x & 7;
  const int dir = blockIdx.y;
  const int wave = threadIdx.x >> 5;
  const int lane = threadIdx.x & 31;

  const unsigned short* Wp = wih_p + (size_t)dir * NP * DP;
  const float* biasp = bias_p + (size_t)dir * NP;
  unsigned int* Gt = gpre_u + (((size_t)dir * TT + t) * 8 + bc) * NTIL * 128;

  // A-matrix address (16-bit A 16x32 layout); row = batch bc*16+m at time t
  const int m = lane & 15;
  const int b = bc * 16 + m;
  int tsrc = t;
  if (dir) { int tr = len[b] - 1 - t; tsrc = tr < 0 ? 0 : (tr > TT - 1 ? TT - 1 : tr); }
  const unsigned int* xrow = (const unsigned int*)xbf + ((size_t)(b * TT + tsrc) * DP >> 1);
  const int kbA = (lane < 16) ? 0 : 8;
  const int kbB = (lane < 16) ? 0 : 8;

  v16bf A[10];
#pragma unroll
  for (int kk = 0; kk < 10; ++kk) {
    v8u au;
#pragma unroll
    for (int j = 0; j < 8; ++j) {
      int K = kk * 32 + kbA + 2 * j + ((j >= 4) ? 8 : 0);
      au[j] = xrow[K >> 1];
    }
    A[kk] = __builtin_bit_cast(v16bf, au);
  }

  for (int nt = wave; nt < NTIL; nt += 8) {
    const int n = nt * 16 + (lane & 15);
    const unsigned int* wrow = (const unsigned int*)Wp + ((size_t)n * DP >> 1);
    if (nt + 8 < NTIL) __builtin_prefetch(wrow + 128 * (DP >> 1), 0, 1);
    float bv = biasp[n];
    v8f acc;
#pragma unroll
    for (int vr = 0; vr < 8; ++vr) acc[vr] = bv;
#pragma unroll
    for (int kk = 0; kk < 10; ++kk) {
      v8u bu;
#pragma unroll
      for (int j = 0; j < 8; ++j) {
        int K = kk * 32 + kbB + 2 * j + ((j >= 4) ? 8 : 0);
        bu[j] = wrow[K >> 1];
      }
      v16bf Bm = __builtin_bit_cast(v16bf, bu);
      acc = __builtin_amdgcn_wmma_f32_16x16x32_bf16(false, A[kk], false, Bm,
                                                    (short)0, acc, false, false);
    }
    // pack 8 accumulators -> 4 dwords of bf16 pairs, single b128 store
    v4u pk;
#pragma unroll
    for (int w = 0; w < 4; ++w)
      pk[w] = (unsigned int)f2bf(acc[2 * w]) | ((unsigned int)f2bf(acc[2 * w + 1]) << 16);
    *(v4u*)(Gt + (size_t)nt * 128 + lane * 4) = pk;
  }
}

// ---------------------------------------------------------------------------
// 4) Recurrent LSTM: grid = 16 blocks (dir = bx>>3, batch-chunk = bx&7), 256 thr
//    Whh (194.5 KB bf16) + h (bf16) + c (f32) all resident in LDS (~210 KB).
__global__ void k_lstm(const unsigned int* __restrict__ gpre_u,
                       const unsigned short* __restrict__ whh_p,
                       unsigned short* __restrict__ hbuf) {
  const int dir = blockIdx.x >> 3;
  const int bc  = blockIdx.x & 7;
  const int wave = threadIdx.x >> 5;
  const int lane = threadIdx.x & 31;

  __shared__ v4u          wsh4[NP * HP / 8];      // 194560 B: Whh in LDS
  __shared__ unsigned int hsh_u[16 * (HP / 2)];   // 16 x 160 bf16 h-state
  __shared__ float        csh[16 * HP];           // 16 x 160 f32 c-state
  unsigned short* hsh_h = (unsigned short*)hsh_u;
  const unsigned int* wsh_u = (const unsigned int*)wsh4;

  // one-time Whh preload (b128 copies)
  {
    const v4u* src = (const v4u*)(whh_p + (size_t)dir * NP * HP);
    for (int i = threadIdx.x; i < NP * HP / 8; i += blockDim.x) wsh4[i] = src[i];
  }
  for (int i = threadIdx.x; i < 16 * (HP / 2); i += blockDim.x) hsh_u[i] = 0u;
  for (int i = threadIdx.x; i < 16 * HP; i += blockDim.x) csh[i] = 0.f;
  __syncthreads();

  const unsigned int* Gp = gpre_u + (size_t)dir * TT * 8 * NTIL * 128;
  unsigned short* Hout = hbuf + (size_t)dir * ROWS * HP;

  const int m     = lane & 15;
  const int kbA   = (lane < 16) ? 0 : 8;
  const int kbB   = (lane < 16) ? 0 : 8;
  const int col   = lane & 15;
  const int rext  = (lane >> 4) << 3;
  const int gbase = lane & ~3;          // lane group holding i,f,g,o of one (b,unit)

  for (int t = 0; t < TT; ++t) {
    // load A k-tiles (K=160 -> 5 tiles) from LDS h
    v16bf A[5];
    const unsigned int* hrow = hsh_u + m * (HP / 2);
#pragma unroll
    for (int kk = 0; kk < 5; ++kk) {
      v8u au;
#pragma unroll
      for (int j = 0; j < 8; ++j) {
        int K = kk * 32 + kbA + 2 * j + ((j >= 4) ? 8 : 0);
        au[j] = hrow[K >> 1];
      }
      A[kk] = __builtin_bit_cast(v16bf, au);
    }
    __syncthreads();   // all A reads done before anyone rewrites h

    for (int nt = wave; nt < NTIL; nt += 8) {
      const int n = nt * 16 + col;
      // accumulator init: one b128 load of packed Gpre tile
      const unsigned int* gp = Gp + (((size_t)t * 8 + bc) * NTIL + nt) * 128 + lane * 4;
      __builtin_prefetch(gp + 8 * NTIL * 128, 0, 1);  // next time-step's tile
      v4u gv = *(const v4u*)gp;
      v8f acc;
#pragma unroll
      for (int w = 0; w < 4; ++w) {
        acc[2 * w]     = bf2f(gv[w]);
        acc[2 * w + 1] = bf2f(gv[w] >> 16);
      }
      const unsigned int* wrow = wsh_u + (size_t)n * (HP / 2);
#pragma unroll
      for (int kk = 0; kk < 5; ++kk) {
        v8u bu;
#pragma unroll
        for (int j = 0; j < 8; ++j) {
          int K = kk * 32 + kbB + 2 * j + ((j >= 4) ? 8 : 0);
          bu[j] = wrow[K >> 1];
        }
        v16bf Bm = __builtin_bit_cast(v16bf, bu);
        acc = __builtin_amdgcn_wmma_f32_16x16x32_bf16(false, A[kk], false, Bm,
                                                      (short)0, acc, false, false);
      }
      // LSTM nonlinearity: 4 adjacent lanes hold i,f,g,o of one (b,unit)
      const int unit = nt * 4 + (col >> 2);
#pragma unroll
      for (int vr = 0; vr < 8; ++vr) {
        float g  = acc[vr];
        float gi = __shfl(g, gbase + 0, 32);
        float gf = __shfl(g, gbase + 1, 32);
        float gg = __shfl(g, gbase + 2, 32);
        float go = __shfl(g, gbase + 3, 32);
        if ((lane & 3) == 0) {
          int br = vr + rext;                 // local batch row 0..15
          float c = csh[br * HP + unit];
          c = sigf(gf) * c + sigf(gi) * tanh_fast(gg);
          csh[br * HP + unit] = c;
          float h = sigf(go) * tanh_fast(c);
          unsigned short hb = f2bf(h);
          hsh_h[br * HP + unit] = hb;
          int bg = bc * 16 + br;
          Hout[(size_t)(bg * TT + t) * HP + unit] = hb;
        }
      }
    }
    __syncthreads();   // new h visible before next step's A loads
  }
}

// ---------------------------------------------------------------------------
// 5) Emissions: out[b,t,l] = (hf . Wl[:, :H] + hb_unrev . Wl[:, H:] + bl) * mask
__global__ void k_emissions(const unsigned short* __restrict__ hbuf,
                            const int* __restrict__ len,
                            const float* __restrict__ Wl,
                            const float* __restrict__ bl,
                            const float* __restrict__ mask,
                            float* __restrict__ emis) {
  int r = blockIdx.x;
  int l = threadIdx.x;
  if (l >= LL) return;
  int b = r >> 9, t = r & (TT - 1);
  int tt = len[b] - 1 - t;
  tt = tt < 0 ? 0 : (tt > TT - 1 ? TT - 1 : tt);
  const unsigned int* hf = (const unsigned int*)(hbuf + (size_t)r * HP);
  const unsigned int* hb = (const unsigned int*)(hbuf + (size_t)ROWS * HP +
                                                 (size_t)(b * TT + tt) * HP);
  const float* wl = Wl + l * DD;
  float s = bl[l];
  for (int k2 = 0; k2 < HH / 2; ++k2) {
    unsigned int w = hf[k2];
    s += bf2f(w) * wl[2 * k2] + bf2f(w >> 16) * wl[2 * k2 + 1];
  }
  for (int k2 = 0; k2 < HH / 2; ++k2) {
    unsigned int w = hb[k2];
    s += bf2f(w) * wl[HH + 2 * k2] + bf2f(w >> 16) * wl[HH + 2 * k2 + 1];
  }
  emis[(size_t)r * LL + l] = s * mask[r];
}

// 6) CRF numerator (one thread per batch row)
__global__ void k_crf_num(const float* __restrict__ emis, const int* __restrict__ labels,
                          const float* __restrict__ mask, const float* __restrict__ trans,
                          const float* __restrict__ strans, const float* __restrict__ etrans,
                          const int* __restrict__ len, float* __restrict__ snum) {
  int b = threadIdx.x;
  if (b >= BB) return;
  const int* lab = labels + b * TT;
  const float* mk = mask + b * TT;
  float s = strans[lab[0]];
  for (int t = 0; t < TT; ++t) s += emis[(size_t)(b * TT + t) * LL + lab[t]] * mk[t];
  int prev = lab[0];
  for (int t = 1; t < TT; ++t) { int cur = lab[t]; s += trans[prev * LL + cur] * mk[t]; prev = cur; }
  s += etrans[lab[len[b] - 1]];
  snum[b] = s;
}

// 7) CRF denominator: forward algorithm, one wave per batch row, lane = label
__global__ void k_crf_den(const float* __restrict__ emis, const float* __restrict__ mask,
                          const float* __restrict__ trans, const float* __restrict__ strans,
                          const float* __restrict__ etrans, float* __restrict__ sden) {
  int b = blockIdx.x, l = threadIdx.x;
  __shared__ float alpha[LL], nbuf[LL];
  if (l < LL) alpha[l] = strans[l] + emis[(size_t)(b * TT) * LL + l];
  __syncthreads();
  for (int t = 1; t < TT; ++t) {
    float mv = mask[b * TT + t];
    if (mv > 0.f && l < LL) {
      float mx = -1e30f;
      for (int i = 0; i < LL; ++i) { float v = alpha[i] + trans[i * LL + l]; mx = v > mx ? v : mx; }
      float sum = 0.f;
      for (int i = 0; i < LL; ++i) sum += __expf(alpha[i] + trans[i * LL + l] - mx);
      nbuf[l] = mx + __logf(sum) + emis[(size_t)(b * TT + t) * LL + l];
    }
    __syncthreads();
    if (mv > 0.f && l < LL) alpha[l] = nbuf[l];
    __syncthreads();
  }
  if (l == 0) {
    float mx = -1e30f;
    for (int i = 0; i < LL; ++i) { float v = alpha[i] + etrans[i]; mx = v > mx ? v : mx; }
    float sum = 0.f;
    for (int i = 0; i < LL; ++i) sum += __expf(alpha[i] + etrans[i] - mx);
    sden[b] = mx + __logf(sum);
  }
}

// 8) Viterbi + backtrace; writes path (as float) into d_out[1:]
__global__ void k_viterbi(const float* __restrict__ emis, const float* __restrict__ mask,
                          const float* __restrict__ trans, const float* __restrict__ strans,
                          const float* __restrict__ etrans, unsigned char* __restrict__ bp,
                          float* __restrict__ out) {
  int b = blockIdx.x, l = threadIdx.x;
  __shared__ float delta[LL], nd[LL];
  if (l < LL) delta[l] = strans[l] + emis[(size_t)(b * TT) * LL + l];
  __syncthreads();
  for (int t = 1; t < TT; ++t) {
    float mv = mask[b * TT + t];
    if (l < LL) {
      if (mv > 0.f) {
        float best = -1e30f; int bi = 0;
        for (int i = 0; i < LL; ++i) {
          float v = delta[i] + trans[i * LL + l];
          if (v > best) { best = v; bi = i; }
        }
        nd[l] = best + emis[(size_t)(b * TT + t) * LL + l];
        bp[(size_t)(b * TT + t) * LL + l] = (unsigned char)bi;
      } else {
        nd[l] = delta[l];
        bp[(size_t)(b * TT + t) * LL + l] = (unsigned char)l;
      }
    }
    __syncthreads();
    if (l < LL) delta[l] = nd[l];
    __syncthreads();
  }
  if (l == 0) {
    float best = -1e30f; int last = 0;
    for (int i = 0; i < LL; ++i) {
      float v = delta[i] + etrans[i];
      if (v > best) { best = v; last = i; }
    }
    float* path = out + 1 + (size_t)b * TT;
    int labc = last;
    path[TT - 1] = (float)labc;
    for (int t = TT - 1; t >= 1; --t) {
      labc = bp[(size_t)(b * TT + t) * LL + labc];
      path[t - 1] = (float)labc;
    }
  }
}

// 9) loss = -mean(num - den)
__global__ void k_loss(const float* __restrict__ snum, const float* __restrict__ sden,
                       float* __restrict__ out) {
  __shared__ float red[BB];
  int b = threadIdx.x;
  if (b < BB) red[b] = snum[b] - sden[b];
  __syncthreads();
  if (b == 0) {
    float s = 0.f;
    for (int i = 0; i < BB; ++i) s += red[i];
    out[0] = -s / (float)BB;
  }
}

// ---------------------------------------------------------------------------
extern "C" void kernel_launch(void* const* d_in, const int* in_sizes, int n_in,
                              void* d_out, int out_size, void* d_ws, size_t ws_size,
                              hipStream_t stream) {
  const float* input_embed = (const float*)d_in[0];
  const float* mask        = (const float*)d_in[1];
  const int*   labels      = (const int*)d_in[2];
  const float* Wih_f = (const float*)d_in[3];
  const float* Whh_f = (const float*)d_in[4];
  const float* bih_f = (const float*)d_in[5];
  const float* bhh_f = (const float*)d_in[6];
  const float* Wih_b = (const float*)d_in[7];
  const float* Whh_b = (const float*)d_in[8];
  const float* bih_b = (const float*)d_in[9];
  const float* bhh_b = (const float*)d_in[10];
  const float* Wl    = (const float*)d_in[11];
  const float* bl    = (const float*)d_in[12];
  const float* trans = (const float*)d_in[13];
  const float* strn  = (const float*)d_in[14];
  const float* etrn  = (const float*)d_in[15];
  float* out = (float*)d_out;

  char* ws = (char*)d_ws;
  size_t off = 0;
  auto alloc = [&](size_t bytes) { size_t r = off; off += (bytes + 255) & ~(size_t)255; return r; };
  int*            lengths = (int*)(ws + alloc(BB * sizeof(int)));
  unsigned short* xbf     = (unsigned short*)(ws + alloc((size_t)ROWS * DP * 2));
  unsigned short* wih_p   = (unsigned short*)(ws + alloc(2ULL * NP * DP * 2));
  unsigned short* whh_p   = (unsigned short*)(ws + alloc(2ULL * NP * HP * 2));
  float*          bias_p  = (float*)(ws + alloc(2ULL * NP * 4));
  unsigned int*   gpre_u  = (unsigned int*)(ws + alloc(2ULL * ROWS * NP * 2));
  unsigned short* hbuf    = (unsigned short*)(ws + alloc(2ULL * ROWS * HP * 2));
  float*          emis    = (float*)(ws + alloc((size_t)ROWS * LL * 4));
  float*          snum    = (float*)(ws + alloc(BB * 4));
  float*          sden    = (float*)(ws + alloc(BB * 4));
  unsigned char*  bp      = (unsigned char*)(ws + alloc((size_t)ROWS * LL));

  k_lengths<<<1, 128, 0, stream>>>(mask, lengths);

  {
    long long total = (long long)ROWS * DP;
    int blocks = (int)((total + 255) / 256);
    k_pack_x<<<blocks, 256, 0, stream>>>(input_embed, xbf);
  }
  k_pack_w<<<1024, 256, 0, stream>>>(Wih_f, Whh_f, bih_f, bhh_f,
                                     Wih_b, Whh_b, bih_b, bhh_b,
                                     wih_p, whh_p, bias_p);

  dim3 g1(TT * 8, 2);
  k_gemm_gpre<<<g1, 256, 0, stream>>>(xbf, wih_p, bias_p, lengths, gpre_u);

  k_lstm<<<16, 256, 0, stream>>>(gpre_u, whh_p, hbuf);

  k_emissions<<<ROWS, 32, 0, stream>>>(hbuf, lengths, Wl, bl, mask, emis);

  k_crf_num<<<1, 128, 0, stream>>>(emis, labels, mask, trans, strn, etrn, lengths, snum);
  k_crf_den<<<BB, 32, 0, stream>>>(emis, mask, trans, strn, etrn, sden);
  k_viterbi<<<BB, 32, 0, stream>>>(emis, mask, trans, strn, etrn, bp, out);
  k_loss<<<1, 128, 0, stream>>>(snum, sden, out);
}
